// GRUSAIE_26903675142845
// MI455X (gfx1250) — matrix operand order repaired and checked by reading, not compiled, not run on hardware
//
#include <hip/hip_runtime.h>
#include <hip/hip_bf16.h>
#include <math.h>

// ---------------------------------------------------------------------------
// Problem dims (compile-time, from the reference)
// ---------------------------------------------------------------------------
#define B_ 8
#define S_ 1024
#define V_ 50000
#define E_ 256
#define H_ 256
#define H2_ 512
#define C_ 64
#define G_ 768        // 3*H
#define G2_ 1536      // 3*H2
#define BS_ 8192      // B*S

typedef __attribute__((ext_vector_type(16))) __bf16 v16bf;
typedef __attribute__((ext_vector_type(8)))  float  v8f;
typedef __attribute__((ext_vector_type(4)))  unsigned int u32x4;
typedef __attribute__((ext_vector_type(8)))  unsigned int u32x8;
typedef unsigned short ushort_t;

// ---------------------------------------------------------------------------
// Helpers
// ---------------------------------------------------------------------------
static __device__ __forceinline__ ushort_t f32_to_bf16(float f) {
  unsigned int u = __float_as_uint(f);
  u += 0x7FFFu + ((u >> 16) & 1u);          // RNE
  return (ushort_t)(u >> 16);
}

static __device__ __forceinline__ float sigmoid_(float x) {
  return 1.0f / (1.0f + __expf(-x));
}

// Low 32 bits of a generic pointer to LDS == LDS byte offset (flat aperture).
static __device__ __forceinline__ unsigned lds_off32(const void* p) {
  return (unsigned)(unsigned long long)p;
}

// Load a 16-element bf16 fragment as two 16-byte chunks (A-frag: p1 = p0+8
// elems; B-frag: p1 = p0+8 elems, i.e. one contiguous 32-byte run).
static __device__ __forceinline__ v16bf ld_frag(const ushort_t* p0,
                                                const ushort_t* p1) {
  union { uint4 q[2]; v16bf v; } u;
  u.q[0] = *reinterpret_cast<const uint4*>(p0);
  u.q[1] = *reinterpret_cast<const uint4*>(p1);
  return u.v;
}

static __device__ __forceinline__ v8f wmma_bf16(v16bf a, v16bf b, v8f c) {
  return __builtin_amdgcn_wmma_f32_16x16x32_bf16(
      /*neg_a=*/false, a, /*neg_b=*/false, b,
      /*c_mod=*/(short)0, c, /*reuse_a=*/false, /*reuse_b=*/false);
}

// CDNA5 async global->LDS copy, 16B per lane; tracked with ASYNCcnt.
static __device__ __forceinline__ void async_g2l_b128(unsigned lds_byte_off,
                                                      const void* gaddr) {
  asm volatile("global_load_async_to_lds_b128 %0, %1, off"
               :: "v"(lds_byte_off), "v"(gaddr)
               : "memory");
}

// CDNA5 Tensor Data Mover: 2-D tile (tile_h rows x tile_w elems) from global
// to LDS.  D# group0/group1 per CDNA5 ISA (§8.3/§8.4);
// VADDR2/VADDR3 omitted (<=2-D tensor).  Tracked with TENSORcnt.
static __device__ __forceinline__ void tdm_load_2d(unsigned lds_byte_off,
                                                   const void* gptr,
                                                   unsigned tile_w,     // elems
                                                   unsigned tile_h,     // rows
                                                   unsigned long long row_stride,  // elems
                                                   unsigned data_size_code) { // 0:1B 1:2B 2:4B
  unsigned long long ga = (unsigned long long)gptr;
  u32x4 g0;
  g0[0] = 0x1u;                                           // count=1, user mode
  g0[1] = lds_byte_off;                                   // lds_addr
  g0[2] = (unsigned)ga;                                   // global_addr[31:0]
  g0[3] = (unsigned)((ga >> 32) & 0x01FFFFFFu)            // global_addr[56:32]
          | 0x80000000u;                                  // type=2 ("image")
  u32x8 g1;
  g1[0] = (data_size_code & 3u) << 16;                    // wg_mask=0, data_size
  g1[1] = (tile_w & 0xFFFFu) << 16;                       // tensor_dim0[15:0]
  g1[2] = (tile_w >> 16) | ((tile_h & 0xFFFFu) << 16);    // td0[31:16] | td1[15:0]
  g1[3] = (tile_h >> 16) | ((tile_w & 0xFFFFu) << 16);    // td1[31:16] | tile_dim0
  g1[4] = (tile_h & 0xFFFFu);                             // tile_dim1 (tile_dim2=0)
  g1[5] = (unsigned)(row_stride & 0xFFFFFFFFull);         // dim0_stride[31:0]
  g1[6] = (unsigned)((row_stride >> 32) & 0xFFFFull);     // dim0_stride[47:32]
  g1[7] = 0u;                                             // dim1_stride (unused)
  asm volatile("tensor_load_to_lds %0, %1" :: "s"(g0), "s"(g1) : "memory");
}

// ---------------------------------------------------------------------------
// Elementwise conversion / gather kernels
// ---------------------------------------------------------------------------
__global__ void cvt_f2bf_kernel(const float* __restrict__ src,
                                ushort_t* __restrict__ dst, long long n) {
  long long i = (long long)blockIdx.x * blockDim.x + threadIdx.x;
  if (i < n) dst[i] = f32_to_bf16(src[i]);
}

__global__ void embed_gather_kernel(const int* __restrict__ tok,
                                    const float* __restrict__ emb,
                                    ushort_t* __restrict__ x_bf) {
  long long i = (long long)blockIdx.x * blockDim.x + threadIdx.x;
  if (i >= (long long)BS_ * E_) return;
  long long pos = i >> 8;              // /E_
  int e = (int)(i & (E_ - 1));
  long long row = (long long)tok[pos];
  x_bf[i] = f32_to_bf16(emb[row * E_ + e]);
}

// ---------------------------------------------------------------------------
// Generic batched GEMM:  D[M,N] = A[M,K] * W[N,K]^T (+ bias[N])
// A, W bf16 row-major (leading dim = K), D f32 row-major (leading dim = N).
// Requires: M % 128 == 0, N % 64 == 0, K % 32 == 0 (true for all call sites).
// Block = 256 threads (8 waves).  The 64-row W slab for the block's N-slab is
// staged in LDS by double-buffered async global->LDS copies; B-fragments are
// loaded into independent registers so the 4 WMMAs issue back-to-back.
// ---------------------------------------------------------------------------
__global__ __launch_bounds__(256) void gemm_nt_bf16_kernel(
    const ushort_t* __restrict__ A, const ushort_t* __restrict__ W,
    const float* __restrict__ bias, float* __restrict__ D,
    int M, int N, int K,
    long long sA, long long sW, long long sD) {
  __shared__ ushort_t wbuf[2 * 64 * 32];   // 2 x 4 KB K-slab double buffer

  const int bz = blockIdx.z;
  A += (long long)bz * sA;
  W += (long long)bz * sW;
  D += (long long)bz * sD;

  const int tid  = threadIdx.x;
  const int wave = tid >> 5;
  const int lane = tid & 31;
  const int m0   = blockIdx.y * 128 + wave * 16;
  const int n0b  = blockIdx.x * 64;
  const int mrow = lane & 15;     // M (A) / N (B) index within tile
  const int kb   = lane >> 4;     // K half selector

  // async W staging: thread t copies 8 elems (16 B) of row (t>>2), seg (t&3)
  const int wrow = tid >> 2;      // 0..63
  const int wseg = tid & 3;       // 0..3
  const ushort_t* wsrc = W + (long long)(n0b + wrow) * K + wseg * 8;
  const unsigned my_lds = lds_off32(wbuf) + (unsigned)((wrow * 32 + wseg * 8) * 2);

  v8f acc0 = {}, acc1 = {}, acc2 = {}, acc3 = {};

  auto compute_tile = [&](int k0, int cur) {
    const ushort_t* pa = A + (long long)(m0 + mrow) * K + k0 + kb * 16;
    __builtin_prefetch(pa + 32, 0, 0);                 // next A k-tile
    const v16bf afrag = ld_frag(pa, pa + 8);
    const ushort_t* pb0 = wbuf + cur * 2048 + (mrow * 32 + kb * 16);
    // independent B fragments -> 8 ds_loads, one wait, 4 back-to-back WMMAs
    const v16bf b0 = ld_frag(pb0,        pb0 + 8);
    const v16bf b1 = ld_frag(pb0 + 512,  pb0 + 520);
    const v16bf b2 = ld_frag(pb0 + 1024, pb0 + 1032);
    const v16bf b3 = ld_frag(pb0 + 1536, pb0 + 1544);
    acc0 = wmma_bf16(afrag, b0, acc0);
    acc1 = wmma_bf16(afrag, b1, acc1);
    acc2 = wmma_bf16(afrag, b2, acc2);
    acc3 = wmma_bf16(afrag, b3, acc3);
  };

  const int nsteps = K >> 5;
  async_g2l_b128(my_lds, wsrc);                        // prologue: k-tile 0
  for (int s = 0; s < nsteps - 1; ++s) {               // branch-free steady state
    const int cur = s & 1;
    async_g2l_b128(my_lds + (unsigned)((cur ^ 1) * 4096), wsrc + (s << 5) + 32);
    asm volatile("s_wait_asynccnt 0x1" ::: "memory");  // tile `cur` done
    __syncthreads();                                   // slab visible to all
    compute_tile(s << 5, cur);
    __syncthreads();                                   // done reading `cur`
  }
  asm volatile("s_wait_asynccnt 0x0" ::: "memory");    // epilogue tile
  __syncthreads();
  compute_tile((nsteps - 1) << 5, (nsteps - 1) & 1);

  const int rowsel = kb;  // lanes 0-15 -> rows r, lanes 16-31 -> rows r+8
  v8f accs[4] = {acc0, acc1, acc2, acc3};
#pragma unroll
  for (int j = 0; j < 4; ++j) {
    const int col = n0b + j * 16 + mrow;
    const float bv = bias ? bias[col] : 0.0f;
#pragma unroll
    for (int r = 0; r < 8; ++r) {
      const int row = m0 + r + rowsel * 8;
      D[(long long)row * N + col] = accs[j][r] + bv;
    }
  }
}

// ---------------------------------------------------------------------------
// Bi-directional GRU recurrence. One persistent workgroup per direction
// (gridDim.x == 2).  768 threads = 24 waves; 48 N-tiles of Wh -> 2 per wave,
// computed kt-outer with 2 independent accumulators (A-frag loaded once per
// k-step; 2 interleaved WMMA chains hide the accumulate latency).
// The per-step xg tile (8 x G f32) is TDM-prefetched into a double-buffered
// LDS slab one step ahead (tensor_load_to_lds + s_wait_tensorcnt).
// Dyn-LDS layout: h_bf[16*H] | h32[8*H] | hg[8*G] | xg_lds[2][8*G]
// ---------------------------------------------------------------------------
__global__ __launch_bounds__(768) void bigru_scan_kernel(
    const float* __restrict__ xg_f, const float* __restrict__ xg_b,
    const ushort_t* __restrict__ wh_f, const ushort_t* __restrict__ wh_b,
    const float* __restrict__ bh_f, const float* __restrict__ bh_b,
    float* __restrict__ out_f, float* __restrict__ out_b,
    float* __restrict__ hT_f, float* __restrict__ hT_b) {
  extern __shared__ char smem_raw[];
  ushort_t* h_bf  = (ushort_t*)smem_raw;                            // 16*H bf16
  float*    h32   = (float*)(smem_raw + 16 * H_ * 2);               // 8*H f32
  float*    hg    = (float*)(smem_raw + 16 * H_ * 2 + 8 * H_ * 4);  // 8*G f32
  float*    xgl   = hg + 8 * G_;                                    // 2x 8*G f32

  const int dir = blockIdx.x;
  const float*    xg  = dir ? xg_b : xg_f;
  const ushort_t* wh  = dir ? wh_b : wh_f;
  const float*    bh  = dir ? bh_b : bh_f;
  float*          out = dir ? out_b : out_f;
  float*          hT  = dir ? hT_b : hT_f;

  const int tid  = threadIdx.x;
  const int wave = tid >> 5;
  const int lane = tid & 31;
  const int mrow = lane & 15;
  const int kb   = lane >> 4;
  const int nt0  = wave * 2;         // first of this wave's two N-tiles

  for (int i = tid; i < 16 * H_; i += blockDim.x) h_bf[i] = 0;
  for (int i = tid; i < 8 * H_; i += blockDim.x)  h32[i] = 0.0f;

  // TDM prologue: fetch xg tile for step 0 into buffer 0
  if (wave == 0) {
    const int ti0 = dir ? (S_ - 1) : 0;
    tdm_load_2d(lds_off32(xgl), xg + (long long)ti0 * G_,
                G_, B_, (unsigned long long)S_ * G_, /*4B*/2u);
    __builtin_amdgcn_s_wait_tensorcnt(0);
  }
  __syncthreads();

  for (int step = 0; step < S_; ++step) {
    const int ti  = dir ? (S_ - 1 - step) : step;
    const int cur = step & 1;

    // TDM prefetch of next step's xg tile into the other buffer
    if (wave == 0 && step + 1 < S_) {
      const int tn = dir ? (S_ - 2 - step) : (step + 1);
      tdm_load_2d(lds_off32(xgl + (cur ^ 1) * 8 * G_),
                  xg + (long long)tn * G_,
                  G_, B_, (unsigned long long)S_ * G_, 2u);
    }

    // ---- hg = h @ Wh^T + bh  (WMMA over LDS-resident h) ----
    v8f c0 = {}, c1 = {};
#pragma unroll
    for (int kt = 0; kt < 8; ++kt) {
      const int k0 = kt * 32;
      const ushort_t* pa = h_bf + mrow * H_ + k0 + kb * 16;
      const v16bf afrag = ld_frag(pa, pa + 8);
      const ushort_t* pb0 = wh + (long long)(nt0 * 16 + mrow) * H_ + k0 + kb * 16;
      const ushort_t* pb1 = pb0 + 16 * H_;
      const v16bf b0 = ld_frag(pb0, pb0 + 8);
      const v16bf b1 = ld_frag(pb1, pb1 + 8);
      c0 = wmma_bf16(afrag, b0, c0);
      c1 = wmma_bf16(afrag, b1, c1);
    }
    if (kb == 0) {  // lanes 0-15 hold output rows 0..7 (batch rows)
      const int col0 = nt0 * 16 + mrow;
      const float bv0 = bh[col0], bv1 = bh[col0 + 16];
#pragma unroll
      for (int r = 0; r < 8; ++r) {
        hg[r * G_ + col0]      = c0[r] + bv0;
        hg[r * G_ + col0 + 16] = c1[r] + bv1;
      }
    }
    __syncthreads();

    // ---- gate math + state update (xg tile comes from LDS) ----
    const float* xt = xgl + cur * 8 * G_;
    for (int idx = tid; idx < B_ * H_; idx += blockDim.x) {
      const int bb = idx >> 8;            // / H_
      const int j  = idx & (H_ - 1);
      const float r = sigmoid_(xt[bb * G_ + j]            + hg[bb * G_ + j]);
      const float z = sigmoid_(xt[bb * G_ + H_ + j]       + hg[bb * G_ + H_ + j]);
      const float n = tanhf  (xt[bb * G_ + 2 * H_ + j] + r * hg[bb * G_ + 2 * H_ + j]);
      const float hn = (1.0f - z) * n + z * h32[idx];
      h32[idx] = hn;
      h_bf[bb * H_ + j] = f32_to_bf16(hn);
      out[((long long)bb * S_ + ti) * H_ + j] = hn;
    }
    __syncthreads();

    // make next tile visible to all waves before the next gate phase
    if (wave == 0 && step + 1 < S_) __builtin_amdgcn_s_wait_tensorcnt(0);
    __syncthreads();
  }
  for (int idx = tid; idx < B_ * H_; idx += blockDim.x) hT[idx] = h32[idx];
}

// ---------------------------------------------------------------------------
// Unidirectional GRU recurrence (H2=512). One workgroup, 1024 threads =
// 32 waves; 96 N-tiles -> 3 per wave, kt-outer with 3 independent
// accumulators.  Same TDM xg prefetch scheme.
// Dyn-LDS: h_bf[16*H2] | h32[8*H2] | hg[8*G2] | xg_lds[2][8*G2]
// ---------------------------------------------------------------------------
__global__ __launch_bounds__(1024) void unigru_scan_kernel(
    const float* __restrict__ xg, const ushort_t* __restrict__ wh,
    const float* __restrict__ bh, float* __restrict__ out) {
  extern __shared__ char smem_raw[];
  ushort_t* h_bf = (ushort_t*)smem_raw;                              // 16*H2
  float*    h32  = (float*)(smem_raw + 16 * H2_ * 2);                // 8*H2
  float*    hg   = (float*)(smem_raw + 16 * H2_ * 2 + 8 * H2_ * 4);  // 8*G2
  float*    xgl  = hg + 8 * G2_;                                     // 2x 8*G2

  const int tid  = threadIdx.x;
  const int wave = tid >> 5;
  const int lane = tid & 31;
  const int mrow = lane & 15;
  const int kb   = lane >> 4;
  const int nt0  = wave * 3;         // first of this wave's three N-tiles

  for (int i = tid; i < 16 * H2_; i += blockDim.x) h_bf[i] = 0;
  for (int i = tid; i < 8 * H2_; i += blockDim.x)  h32[i] = 0.0f;

  if (wave == 0) {
    tdm_load_2d(lds_off32(xgl), xg, G2_, B_,
                (unsigned long long)S_ * G2_, 2u);
    __builtin_amdgcn_s_wait_tensorcnt(0);
  }
  __syncthreads();

  for (int ti = 0; ti < S_; ++ti) {
    const int cur = ti & 1;
    if (wave == 0 && ti + 1 < S_) {
      tdm_load_2d(lds_off32(xgl + (cur ^ 1) * 8 * G2_),
                  xg + (long long)(ti + 1) * G2_,
                  G2_, B_, (unsigned long long)S_ * G2_, 2u);
    }

    v8f c0 = {}, c1 = {}, c2 = {};
#pragma unroll
    for (int kt = 0; kt < 16; ++kt) {
      const int k0 = kt * 32;
      const ushort_t* pa = h_bf + mrow * H2_ + k0 + kb * 16;
      const v16bf afrag = ld_frag(pa, pa + 8);
      const ushort_t* pb0 = wh + (long long)(nt0 * 16 + mrow) * H2_ + k0 + kb * 16;
      const ushort_t* pb1 = pb0 + 16 * H2_;
      const ushort_t* pb2 = pb0 + 32 * H2_;
      const v16bf b0 = ld_frag(pb0, pb0 + 8);
      const v16bf b1 = ld_frag(pb1, pb1 + 8);
      const v16bf b2 = ld_frag(pb2, pb2 + 8);
      c0 = wmma_bf16(afrag, b0, c0);
      c1 = wmma_bf16(afrag, b1, c1);
      c2 = wmma_bf16(afrag, b2, c2);
    }
    if (kb == 0) {
      const int col0 = nt0 * 16 + mrow;
      const float bv0 = bh[col0], bv1 = bh[col0 + 16], bv2 = bh[col0 + 32];
#pragma unroll
      for (int r = 0; r < 8; ++r) {
        hg[r * G2_ + col0]      = c0[r] + bv0;
        hg[r * G2_ + col0 + 16] = c1[r] + bv1;
        hg[r * G2_ + col0 + 32] = c2[r] + bv2;
      }
    }
    __syncthreads();

    const float* xt = xgl + cur * 8 * G2_;
    for (int idx = tid; idx < B_ * H2_; idx += blockDim.x) {
      const int bb = idx >> 9;            // / H2_
      const int j  = idx & (H2_ - 1);
      const float r = sigmoid_(xt[bb * G2_ + j]             + hg[bb * G2_ + j]);
      const float z = sigmoid_(xt[bb * G2_ + H2_ + j]       + hg[bb * G2_ + H2_ + j]);
      const float n = tanhf  (xt[bb * G2_ + 2 * H2_ + j] + r * hg[bb * G2_ + 2 * H2_ + j]);
      const float hn = (1.0f - z) * n + z * h32[idx];
      h32[idx] = hn;
      h_bf[bb * H2_ + j] = f32_to_bf16(hn);
      out[((long long)bb * S_ + ti) * H2_ + j] = hn;
    }
    __syncthreads();

    if (wave == 0 && ti + 1 < S_) __builtin_amdgcn_s_wait_tensorcnt(0);
    __syncthreads();
  }
}

// ---------------------------------------------------------------------------
// Assembly / glue kernels
// ---------------------------------------------------------------------------
__global__ void build_bigru_kernel(const float* __restrict__ out_f,
                                   const float* __restrict__ out_b,
                                   float* __restrict__ big32,
                                   ushort_t* __restrict__ big_bf) {
  long long i = (long long)blockIdx.x * blockDim.x + threadIdx.x;
  if (i >= (long long)BS_ * H2_) return;
  const int d = (int)(i & (H2_ - 1));
  const long long ts = i >> 9;  // b*S + t
  const float v = (d < H_) ? out_f[ts * H_ + d] : out_b[ts * H_ + (d - H_)];
  big32[i] = v;
  big_bf[i] = f32_to_bf16(v);
}

__global__ void build_U_kernel(const float* __restrict__ hT_f,
                               const float* __restrict__ hT_b,
                               const float* __restrict__ uni_out,
                               ushort_t* __restrict__ U_bf) {
  long long i = (long long)blockIdx.x * blockDim.x + threadIdx.x;
  if (i >= (long long)BS_ * H2_) return;
  const int d = (int)(i & (H2_ - 1));
  const long long ts = i >> 9;
  const int t = (int)(ts & (S_ - 1));
  const int bb = (int)(ts >> 10);
  float v;
  if (t == 0) {
    v = (d < H_) ? hT_b[bb * H_ + d] : hT_f[bb * H_ + (d - H_)];  // [hT_b, hT_f]
  } else {
    v = uni_out[((long long)bb * S_ + (t - 1)) * H2_ + d];
  }
  U_bf[i] = f32_to_bf16(v);
}

__global__ void transpose_bf_kernel(const ushort_t* __restrict__ big_bf,
                                    ushort_t* __restrict__ bigT_bf) {
  long long i = (long long)blockIdx.x * blockDim.x + threadIdx.x;
  if (i >= (long long)BS_ * H2_) return;
  const int d = (int)(i & (H2_ - 1));
  const long long ts = i >> 9;
  const int t = (int)(ts & (S_ - 1));
  const int bb = (int)(ts >> 10);
  bigT_bf[((long long)bb * H2_ + d) * S_ + t] = big_bf[i];
}

// Row softmax over S_ elements; one block (256 threads) per row; bf16 out.
__global__ __launch_bounds__(256) void softmax_row_kernel(
    const float* __restrict__ energies, ushort_t* __restrict__ probs_bf) {
  __shared__ float red[256];
  const long long row = blockIdx.x;
  const float* e = energies + row * S_;
  const int tid = threadIdx.x;

  float v[4];
  float m = -1e30f;
#pragma unroll
  for (int j = 0; j < 4; ++j) {
    v[j] = e[tid + j * 256];
    m = fmaxf(m, v[j]);
  }
  red[tid] = m;
  __syncthreads();
  for (int s = 128; s > 0; s >>= 1) {
    if (tid < s) red[tid] = fmaxf(red[tid], red[tid + s]);
    __syncthreads();
  }
  m = red[0];
  __syncthreads();

  float sum = 0.0f;
#pragma unroll
  for (int j = 0; j < 4; ++j) {
    v[j] = __expf(v[j] - m);
    sum += v[j];
  }
  red[tid] = sum;
  __syncthreads();
  for (int s = 128; s > 0; s >>= 1) {
    if (tid < s) red[tid] += red[tid + s];
    __syncthreads();
  }
  const float inv = 1.0f / red[0];
#pragma unroll
  for (int j = 0; j < 4; ++j)
    probs_bf[row * S_ + tid + j * 256] = f32_to_bf16(v[j] * inv);
}

__global__ void ft_gate_kernel(const float* __restrict__ cfc,
                               const float* __restrict__ cfcfc,
                               const float* __restrict__ htfc,
                               const float* __restrict__ big32,
                               ushort_t* __restrict__ ft_bf) {
  long long i = (long long)blockIdx.x * blockDim.x + threadIdx.x;
  if (i >= (long long)BS_ * H2_) return;
  const float g = sigmoid_(cfcfc[i] + htfc[i]);
  ft_bf[i] = f32_to_bf16(cfc[i] * g + big32[i]);
}

// ---------------------------------------------------------------------------
// Host-side orchestration
// ---------------------------------------------------------------------------
static inline void launch_gemm(const ushort_t* A, const ushort_t* W,
                               const float* bias, float* D, int M, int N,
                               int K, long long sA, long long sW, long long sD,
                               int batch, hipStream_t stream) {
  dim3 grid((N + 63) / 64, (M + 127) / 128, batch);
  gemm_nt_bf16_kernel<<<grid, 256, 0, stream>>>(A, W, bias, D, M, N, K, sA, sW, sD);
}

static inline void launch_cvt(const float* s, ushort_t* d, long long n,
                              hipStream_t stream) {
  cvt_f2bf_kernel<<<(unsigned)((n + 255) / 256), 256, 0, stream>>>(s, d, n);
}

extern "C" void kernel_launch(void* const* d_in, const int* in_sizes, int n_in,
                              void* d_out, int out_size, void* d_ws,
                              size_t ws_size, hipStream_t stream) {
  (void)in_sizes; (void)n_in; (void)out_size; (void)ws_size;

  const int*   padded   = (const int*)d_in[0];
  const float* emb      = (const float*)d_in[2];
  const float* wi_f     = (const float*)d_in[3];
  const float* wh_f     = (const float*)d_in[4];
  const float* bi_f     = (const float*)d_in[5];
  const float* bh_f     = (const float*)d_in[6];
  const float* wi_b     = (const float*)d_in[7];
  const float* wh_b     = (const float*)d_in[8];
  const float* bi_b     = (const float*)d_in[9];
  const float* bh_b     = (const float*)d_in[10];
  const float* uwi      = (const float*)d_in[11];
  const float* uwh      = (const float*)d_in[12];
  const float* ubi      = (const float*)d_in[13];
  const float* ubh      = (const float*)d_in[14];
  const float* w_attn   = (const float*)d_in[15];
  const float* b_attn   = (const float*)d_in[16];
  const float* w_attnfc = (const float*)d_in[17];
  const float* b_attnfc = (const float*)d_in[18];
  const float* w_ht     = (const float*)d_in[19];
  const float* b_ht     = (const float*)d_in[20];
  const float* w_out    = (const float*)d_in[21];
  const float* b_out    = (const float*)d_in[22];
  float* out            = (float*)d_out;

  // ---- workspace carve-up (deterministic bump allocator) ----
  size_t off = 0;
  auto alloc = [&](size_t bytes) -> void* {
    void* p = (char*)d_ws + off;
    off = (off + bytes + 255) & ~(size_t)255;
    return p;
  };
  ushort_t* x_bf     = (ushort_t*)alloc((size_t)BS_ * E_ * 2);
  ushort_t* wif_bf   = (ushort_t*)alloc((size_t)G_ * E_ * 2);
  ushort_t* whf_bf   = (ushort_t*)alloc((size_t)G_ * H_ * 2);
  ushort_t* wib_bf   = (ushort_t*)alloc((size_t)G_ * E_ * 2);
  ushort_t* whb_bf   = (ushort_t*)alloc((size_t)G_ * H_ * 2);
  ushort_t* uwi_bf   = (ushort_t*)alloc((size_t)G2_ * H2_ * 2);
  ushort_t* uwh_bf   = (ushort_t*)alloc((size_t)G2_ * H2_ * 2);
  ushort_t* wat_bf   = (ushort_t*)alloc((size_t)H2_ * H2_ * 2);
  ushort_t* watfc_bf = (ushort_t*)alloc((size_t)H2_ * H2_ * 2);
  ushort_t* wht_bf   = (ushort_t*)alloc((size_t)H2_ * H2_ * 2);
  ushort_t* wout_bf  = (ushort_t*)alloc((size_t)C_ * H2_ * 2);
  float* xg_f        = (float*)alloc((size_t)BS_ * G_ * 4);
  float* xg_b        = (float*)alloc((size_t)BS_ * G_ * 4);
  float* out_f       = (float*)alloc((size_t)BS_ * H_ * 4);
  float* out_b       = (float*)alloc((size_t)BS_ * H_ * 4);
  float* hT_f        = (float*)alloc((size_t)B_ * H_ * 4);
  float* hT_b        = (float*)alloc((size_t)B_ * H_ * 4);
  float* big32       = (float*)alloc((size_t)BS_ * H2_ * 4);
  ushort_t* big_bf   = (ushort_t*)alloc((size_t)BS_ * H2_ * 2);
  float* xg_u        = (float*)alloc((size_t)BS_ * G2_ * 4);
  float* uni_out     = (float*)alloc((size_t)BS_ * H2_ * 4);
  ushort_t* U_bf     = (ushort_t*)alloc((size_t)BS_ * H2_ * 2);
  ushort_t* bigT_bf  = (ushort_t*)alloc((size_t)B_ * H2_ * S_ * 2);
  float* energies    = (float*)alloc((size_t)B_ * S_ * S_ * 4);
  ushort_t* probs_bf = (ushort_t*)alloc((size_t)B_ * S_ * S_ * 2);
  float* ctx32       = (float*)alloc((size_t)BS_ * H2_ * 4);
  ushort_t* ctx_bf   = (ushort_t*)alloc((size_t)BS_ * H2_ * 2);
  float* cfc32       = (float*)alloc((size_t)BS_ * H2_ * 4);
  ushort_t* cfc_bf   = (ushort_t*)alloc((size_t)BS_ * H2_ * 2);
  float* cfcfc32     = (float*)alloc((size_t)BS_ * H2_ * 4);
  float* htfc32      = (float*)alloc((size_t)BS_ * H2_ * 4);
  ushort_t* ft_bf    = (ushort_t*)alloc((size_t)BS_ * H2_ * 2);

  // ---- 1. weight bf16 conversion ----
  launch_cvt(wi_f, wif_bf, (long long)G_ * E_, stream);
  launch_cvt(wh_f, whf_bf, (long long)G_ * H_, stream);
  launch_cvt(wi_b, wib_bf, (long long)G_ * E_, stream);
  launch_cvt(wh_b, whb_bf, (long long)G_ * H_, stream);
  launch_cvt(uwi, uwi_bf, (long long)G2_ * H2_, stream);
  launch_cvt(uwh, uwh_bf, (long long)G2_ * H2_, stream);
  launch_cvt(w_attn, wat_bf, (long long)H2_ * H2_, stream);
  launch_cvt(w_attnfc, watfc_bf, (long long)H2_ * H2_, stream);
  launch_cvt(w_ht, wht_bf, (long long)H2_ * H2_, stream);
  launch_cvt(w_out, wout_bf, (long long)C_ * H2_, stream);

  // ---- 2. embedding gather -> bf16 ----
  {
    long long n = (long long)BS_ * E_;
    embed_gather_kernel<<<(unsigned)((n + 255) / 256), 256, 0, stream>>>(
        padded, emb, x_bf);
  }

  // ---- 3. input-gate GEMMs for both directions ----
  launch_gemm(x_bf, wif_bf, bi_f, xg_f, BS_, G_, E_, 0, 0, 0, 1, stream);
  launch_gemm(x_bf, wib_bf, bi_b, xg_b, BS_, G_, E_, 0, 0, 0, 1, stream);

  // ---- 4. bi-GRU recurrence (fwd + bwd concurrently, one WG each) ----
  {
    size_t smem = (size_t)16 * H_ * 2 + 8 * H_ * 4 + 8 * G_ * 4
                + (size_t)2 * 8 * G_ * 4;  // 88 KB
    bigru_scan_kernel<<<2, 768, smem, stream>>>(
        xg_f, xg_b, whf_bf, whb_bf, bh_f, bh_b, out_f, out_b, hT_f, hT_b);
  }

  // ---- 5. concat -> bigru (f32 + bf16) ----
  {
    long long n = (long long)BS_ * H2_;
    build_bigru_kernel<<<(unsigned)((n + 255) / 256), 256, 0, stream>>>(
        out_f, out_b, big32, big_bf);
  }

  // ---- 6. uni-GRU input gates + recurrence ----
  launch_gemm(big_bf, uwi_bf, ubi, xg_u, BS_, G2_, H2_, 0, 0, 0, 1, stream);
  {
    size_t smem = (size_t)16 * H2_ * 2 + 8 * H2_ * 4 + 8 * G2_ * 4
                + (size_t)2 * 8 * G2_ * 4;  // 176 KB
    unigru_scan_kernel<<<1, 1024, smem, stream>>>(xg_u, uwh_bf, ubh, uni_out);
  }

  // ---- 7. attention ----
  {
    long long n = (long long)BS_ * H2_;
    build_U_kernel<<<(unsigned)((n + 255) / 256), 256, 0, stream>>>(
        hT_f, hT_b, uni_out, U_bf);
    transpose_bf_kernel<<<(unsigned)((n + 255) / 256), 256, 0, stream>>>(
        big_bf, bigT_bf);
  }
  // energies[b] = U[b] (S x H2) * bigru[b]^T  -> (S x S)
  launch_gemm(U_bf, big_bf, nullptr, energies, S_, S_, H2_,
              (long long)S_ * H2_, (long long)S_ * H2_, (long long)S_ * S_,
              B_, stream);
  softmax_row_kernel<<<BS_, 256, 0, stream>>>(energies, probs_bf);
  // context[b] = probs[b] (S x S) * bigru[b]  -> (S x H2) via bigru^T rows
  launch_gemm(probs_bf, bigT_bf, nullptr, ctx32, S_, H2_, S_,
              (long long)S_ * S_, (long long)H2_ * S_, (long long)S_ * H2_,
              B_, stream);
  launch_cvt(ctx32, ctx_bf, (long long)BS_ * H2_, stream);

  // ---- 8. gated FC stack ----
  launch_gemm(ctx_bf, wat_bf, b_attn, cfc32, BS_, H2_, H2_, 0, 0, 0, 1, stream);
  launch_cvt(cfc32, cfc_bf, (long long)BS_ * H2_, stream);
  launch_gemm(cfc_bf, watfc_bf, b_attnfc, cfcfc32, BS_, H2_, H2_, 0, 0, 0, 1, stream);
  launch_gemm(big_bf, wht_bf, b_ht, htfc32, BS_, H2_, H2_, 0, 0, 0, 1, stream);
  {
    long long n = (long long)BS_ * H2_;
    ft_gate_kernel<<<(unsigned)((n + 255) / 256), 256, 0, stream>>>(
        cfc32, cfcfc32, htfc32, big32, ft_bf);
  }

  // ---- 9. output projection -> d_out ----
  launch_gemm(ft_bf, wout_bf, b_out, out, BS_, C_, H2_, 0, 0, 0, 1, stream);
}
